// Encoder_10368051052546
// MI455X (gfx1250) — matrix operand order repaired and checked by reading, not compiled, not run on hardware
//
#include <hip/hip_runtime.h>
#include <hip/hip_bf16.h>

#define VOC   32000
#define EMB_D 512
#define HU    1024          // UNITS
#define BB    128
#define TT    256
#define NG    4096          // 4*HU
#define KTOT  1536          // HU + EMB_D (A = [h | xe])
#define KT_N  48            // KTOT / 32
#define NSPLIT 8            // column-slice blocks per batch group
#define GROUPS 8            // batch groups of 16 rows
#define THREADS 512         // 16 waves

typedef __attribute__((ext_vector_type(16))) __bf16 v16bf;
typedef __attribute__((ext_vector_type(8)))  __bf16 v8bf;
typedef __attribute__((ext_vector_type(8)))  float  v8f;

// ---------------------------------------------------------------------------
// Kernel 0a: build combined, gate-interleaved, WMMA-B-fragment-swizzled
// bf16 weights.  Combined K: rows 0..1023 = U, rows 1024..1535 = W.
// Gate interleave: newcol n = 4*j + gate  <->  origcol = gate*HU + j.
// B-frag swizzle: frag (nt, kt) -> 512 contiguous halves; lane L holds
// N = nt*16 + (L%16), K = kt*32 + (L/16)*16 + e  (e = 0..15 contiguous).
// ---------------------------------------------------------------------------
__global__ void build_weights(const float* __restrict__ W,
                              const float* __restrict__ U,
                              const float* __restrict__ b,
                              __bf16* __restrict__ cw,
                              float* __restrict__ br) {
    size_t idx = (size_t)blockIdx.x * blockDim.x + threadIdx.x;
    if (idx >= (size_t)KTOT * NG) return;
    int k = (int)(idx >> 12);          // 0..1535
    int n = (int)(idx & (NG - 1));     // new (interleaved) column
    int gate = n & 3;
    int j    = n >> 2;
    float v = (k < HU) ? U[(size_t)k * NG + gate * HU + j]
                       : W[(size_t)(k - HU) * NG + gate * HU + j];
    int nt  = n >> 4,  l16 = n & 15;
    int kt  = k >> 5,  kin = k & 31;
    int lane = l16 + ((kin >> 4) << 4);
    int e    = kin & 15;
    cw[((((size_t)nt * KT_N) + kt) << 9) + (lane << 4) + e] = (__bf16)v;
    if (idx < NG) br[idx] = b[(idx & 3) * HU + (idx >> 2)];
}

__global__ void zero_counters(int* __restrict__ cnt) {
    if (threadIdx.x < GROUPS) cnt[threadIdx.x] = 0;
}

// quad broadcast helper: group-of-4 ds_swizzle, sel = 0x00/0x55/0xAA/0xFF
#define QUAD_BCAST(v, SEL) \
    __int_as_float(__builtin_amdgcn_ds_swizzle(__float_as_int(v), 0x8000 | (SEL)))

// ---------------------------------------------------------------------------
// Kernel 1: persistent fused LSTM.
// grid = (NSPLIT, GROUPS), block = 512 threads (16 waves).
// Block (s, g): batch rows [16g,16g+16), interleaved cols [512s, 512s+512).
// Each wave: 2 output tiles of 16x16, K-loop of 48 x (16x16x32 bf16 WMMA).
// h published per step as bf16 via double-buffered global hbuf + per-group
// device-scope atomic barrier.
// ---------------------------------------------------------------------------
__global__ __launch_bounds__(THREADS)
void lstm_scan(const int*   __restrict__ x,
               const float* __restrict__ emb,
               const __bf16* __restrict__ cw,
               const float* __restrict__ br,
               __bf16* __restrict__ hbuf,     // [2][GROUPS][16][1024]
               int*    __restrict__ cnt,      // [GROUPS]
               float*  __restrict__ out,      // [B][T][H]
               float*  __restrict__ hT,       // [B][H]
               float*  __restrict__ cT) {     // [B][H]
    constexpr int APITCH = KTOT + 8;          // 1544 halves -> 3088B row stride
    __shared__ __bf16 A[16 * APITCH];

    const int s    = blockIdx.x;
    const int g    = blockIdx.y;
    const int tid  = threadIdx.x;
    const int wave = tid >> 5;
    const int lane = tid & 31;
    const int lh   = lane >> 4;
    const int l16  = lane & 15;
    const int b0   = g << 4;
    const int ntg0 = (s << 5) + (wave << 1);  // first of 2 N-tiles for wave
    const int gate = lane & 3;
    const bool isG0 = (gate == 0);

    // branch-free activation constants (per-lane):
    //   act(z) = smul * rcp(1 + exp2(cmul*z)) + soff
    //   gate 0,1,3 (i,f,o): sigmoid  -> cmul=-log2e,  smul=1, soff=0
    //   gate 2       (g)  : tanh     -> cmul=-2log2e, smul=2, soff=-1
    const float LOG2E = 1.4426950408889634f;
    const float cmul = (gate == 2) ? -2.0f * LOG2E : -LOG2E;
    const float smul = (gate == 2) ? 2.0f : 1.0f;
    const float soff = (gate == 2) ? -1.0f : 0.0f;

    float creg[2][8];                         // c state (valid in gate-0 lanes)
#pragma unroll
    for (int t2 = 0; t2 < 2; ++t2)
#pragma unroll
        for (int v = 0; v < 8; ++v) creg[t2][v] = 0.0f;

    const float brv0 = br[(ntg0 << 4) + l16];
    const float brv1 = br[((ntg0 + 1) << 4) + l16];
    const int   row  = (lh << 3);             // row base for this lane half
    const int   jq0  = (ntg0 << 2) + (l16 >> 2);
    const int   jq1  = ((ntg0 + 1) << 2) + (l16 >> 2);

    for (int t = 0; t < TT; ++t) {
        // ---- stage A = [h | xe] into LDS ----
        if (t > 0) {
            const __bf16* hr = hbuf + ((size_t)((t & 1) * GROUPS + g) << 14);
#pragma unroll
            for (int i = 0; i < 4; ++i) {
                int c  = (tid << 2) + i;          // 0..2047 chunks of 8 halves
                int r  = c >> 7;
                int c8 = (c & 127) << 3;
                *(v8bf*)&A[r * APITCH + c8] = *(const v8bf*)(hr + (r << 10) + c8);
            }
        }
#pragma unroll
        for (int i = 0; i < 2; ++i) {
            int c  = (tid << 1) + i;              // 0..1023 chunks of 8 floats
            int r  = c >> 6;
            int c8 = (c & 63) << 3;
            int token = x[(b0 + r) * TT + t];
            const float* src = emb + (size_t)token * EMB_D + c8;
            float4 f0 = *(const float4*)(src);
            float4 f1 = *(const float4*)(src + 4);
            v8bf o;
            o[0] = (__bf16)f0.x; o[1] = (__bf16)f0.y;
            o[2] = (__bf16)f0.z; o[3] = (__bf16)f0.w;
            o[4] = (__bf16)f1.x; o[5] = (__bf16)f1.y;
            o[6] = (__bf16)f1.z; o[7] = (__bf16)f1.w;
            *(v8bf*)&A[r * APITCH + HU + c8] = o;
        }
        __syncthreads();

        // ---- GEMM: acc = bias + A(16xK) * CW(Kx32-slice) ----
        v8f acc[2];
#pragma unroll
        for (int e = 0; e < 8; ++e) { acc[0][e] = brv0; acc[1][e] = brv1; }

        const int kstart = (t == 0) ? (HU >> 5) : 0;   // h==0 at t=0: skip U part
        for (int kt = kstart; kt < KT_N; ++kt) {
            const v8bf lo = *(const v8bf*)&A[l16 * APITCH + (kt << 5) + (lh << 4)];
            const v8bf hi = *(const v8bf*)&A[l16 * APITCH + (kt << 5) + (lh << 4) + 8];
            v16bf a;
#pragma unroll
            for (int e = 0; e < 8; ++e) { a[e] = lo[e]; a[8 + e] = hi[e]; }
            const v16bf bm0 =
                *(const v16bf*)(cw + ((((size_t)ntg0)     * KT_N + kt) << 9) + (lane << 4));
            const v16bf bm1 =
                *(const v16bf*)(cw + ((((size_t)ntg0 + 1) * KT_N + kt) << 9) + (lane << 4));
            acc[0] = __builtin_amdgcn_wmma_f32_16x16x32_bf16(
                false, a, false, bm0, (short)0, acc[0], false, false);
            acc[1] = __builtin_amdgcn_wmma_f32_16x16x32_bf16(
                false, a, false, bm1, (short)0, acc[1], false, false);
        }

        // ---- gates, phased for TRANS-latency hiding ----
        // Phase A: 16 independent activations (v_exp_f32 back-to-back)
        float act[2][8];
#pragma unroll
        for (int t2 = 0; t2 < 2; ++t2)
#pragma unroll
            for (int v = 0; v < 8; ++v)
                act[t2][v] = smul * __builtin_amdgcn_rcpf(
                                 1.0f + __builtin_amdgcn_exp2f(cmul * acc[t2][v])) + soff;

        // Phase B: quad exchange (i,f,g,o) + c update; keep o for phase C
        float ogate[2][8];
#pragma unroll
        for (int t2 = 0; t2 < 2; ++t2)
#pragma unroll
            for (int v = 0; v < 8; ++v) {
                float a0 = act[t2][v];
                float ai = QUAD_BCAST(a0, 0x00);
                float af = QUAD_BCAST(a0, 0x55);
                float ag = QUAD_BCAST(a0, 0xAA);
                float ao = QUAD_BCAST(a0, 0xFF);
                // only gate-0 lanes hold live c; others compute junk harmlessly
                creg[t2][v] = af * creg[t2][v] + ai * ag;
                ogate[t2][v] = ao;
            }

        // Phase C: 16 independent tanh(c) evaluations
        float hval[2][8];
#pragma unroll
        for (int t2 = 0; t2 < 2; ++t2)
#pragma unroll
            for (int v = 0; v < 8; ++v) {
                float tc = 2.0f * __builtin_amdgcn_rcpf(
                               1.0f + __builtin_amdgcn_exp2f(
                                   -2.0f * LOG2E * creg[t2][v])) - 1.0f;
                hval[t2][v] = ogate[t2][v] * tc;
            }

        // Phase D: stores (gate-0 lanes only)
        __bf16* hw = hbuf + ((size_t)(((t + 1) & 1) * GROUPS + g) << 14);
        const bool last = (t == TT - 1);
        if (isG0) {
#pragma unroll
            for (int t2 = 0; t2 < 2; ++t2) {
                const int jq = (t2 == 0) ? jq0 : jq1;
#pragma unroll
                for (int v = 0; v < 8; ++v) {
                    float hn = hval[t2][v];
                    int bidx = b0 + row + v;
                    hw[((row + v) << 10) + jq] = (__bf16)hn;
                    out[((size_t)bidx * TT + t) * HU + jq] = hn;
                    if (last) {
                        hT[(size_t)bidx * HU + jq] = hn;
                        cT[(size_t)bidx * HU + jq] = creg[t2][v];
                    }
                }
            }
        }

        // ---- per-group device barrier (release h, then sync, then acquire) ----
        __threadfence();
        __syncthreads();
        if (tid == 0) {
            __hip_atomic_fetch_add(&cnt[g], 1, __ATOMIC_RELEASE,
                                   __HIP_MEMORY_SCOPE_AGENT);
            const int target = NSPLIT * (t + 1);
            while (__hip_atomic_load(&cnt[g], __ATOMIC_ACQUIRE,
                                     __HIP_MEMORY_SCOPE_AGENT) < target)
                __builtin_amdgcn_s_sleep(2);
        }
        __syncthreads();
        __threadfence();
    }
}

// ---------------------------------------------------------------------------
extern "C" void kernel_launch(void* const* d_in, const int* in_sizes, int n_in,
                              void* d_out, int out_size, void* d_ws, size_t ws_size,
                              hipStream_t stream) {
    const int*   x      = (const int*)d_in[0];
    // d_in[1] = hidden (ignored by reference)
    const float* emb    = (const float*)d_in[2];
    const float* W      = (const float*)d_in[3];
    const float* U      = (const float*)d_in[4];
    const float* b      = (const float*)d_in[5];

    float* out = (float*)d_out;                       // [B][T][H]
    float* hT  = out + (size_t)BB * TT * HU;          // [B][H]
    float* cT  = hT + (size_t)BB * HU;                // [B][H]

    // workspace layout
    char* ws = (char*)d_ws;
    __bf16* cw   = (__bf16*)ws;                                   // 12.0 MB
    float*  br   = (float*)(ws + (size_t)KTOT * NG * 2);          // 16 KB
    __bf16* hbuf = (__bf16*)(ws + (size_t)KTOT * NG * 2 + NG * 4);// 512 KB
    int*    cnt  = (int*)((char*)hbuf + (size_t)2 * GROUPS * 16 * HU * 2);

    {   // build swizzled bf16 [U;W] + reordered bias
        size_t total = (size_t)KTOT * NG;
        int blocks = (int)((total + 255) / 256);
        build_weights<<<blocks, 256, 0, stream>>>(W, U, b, cw, br);
    }
    zero_counters<<<1, 32, 0, stream>>>(cnt);

    dim3 grid(NSPLIT, GROUPS);
    lstm_scan<<<grid, THREADS, 0, stream>>>(x, emb, cw, br, hbuf, cnt,
                                            out, hT, cT);
}
